// PartialFC_AdaFace_43233140801637
// MI455X (gfx1250) — matrix operand order repaired and checked by reading, not compiled, not run on hardware
//
#include <hip/hip_runtime.h>
#include <math.h>

// ---------------- problem constants (match reference) ----------------
constexpr int N_ = 1024;       // rows
constexpr int D_ = 512;        // feature dim (K)
constexpr int C_ = 70722;      // classes
constexpr float M_MARGIN = 0.4f;
constexpr float H_SCALE  = 0.333f;
constexpr float S_SCALE  = 64.0f;
constexpr float EPS_     = 0.001f;
constexpr float COS_EPS  = 0.99999950f;   // cos(0.001)

// GEMM tiling
constexpr int CT   = 32;        // classes per block
constexpr int LSTR = 520;       // LDS row stride (elems): 512 + 8 pad -> 4-dword bank step
constexpr int GEMM_BLOCKS = (C_ + CT - 1) / CT;   // 2211

typedef __attribute__((ext_vector_type(16))) __bf16 v16bf;
typedef __attribute__((ext_vector_type(8)))  float  v8f;

// ---------------- bf16 helpers (bit-exact RNE, no scalar __bf16 ops) --------
__device__ __forceinline__ unsigned short f2bf(float f) {
    union { float f; unsigned u; } v; v.f = f;
    unsigned u = v.u;
    unsigned r = u + 0x7FFFu + ((u >> 16) & 1u);
    return (unsigned short)(r >> 16);
}
__device__ __forceinline__ float bf2f(unsigned short h) {
    union { unsigned u; float f; } v; v.u = ((unsigned)h) << 16;
    return v.f;
}

// A fragment: lane holds 8 elems at +0 and 8 elems at +16 (ISA 16-bit A layout)
__device__ __forceinline__ v16bf load_a_frag(const unsigned short* p) {
    union { uint4 q[2]; v16bf v; } u;
    u.q[0] = *reinterpret_cast<const uint4*>(p);
    u.q[1] = *reinterpret_cast<const uint4*>(p + 16);
    return u.v;
}
// B fragment: lane holds 8 elems at +0 and 8 elems at +16 (same K order as A)
__device__ __forceinline__ v16bf load_b_frag(const unsigned short* p) {
    union { uint4 q[2]; v16bf v; } u;
    u.q[0] = *reinterpret_cast<const uint4*>(p);
    u.q[1] = *reinterpret_cast<const uint4*>(p + 16);
    return u.v;
}

__device__ __forceinline__ float wave_sum32(float v) {
    #pragma unroll
    for (int m = 16; m >= 1; m >>= 1) v += __shfl_xor(v, m, 32);
    return v;
}

// ---------------- kernel 1: normalize x -> bf16, safe norms, zero acc -------
__global__ void k_normalize_x(const float* __restrict__ x,
                              unsigned short* __restrict__ xn,
                              float* __restrict__ safe_norms,
                              float* __restrict__ sumexp) {
    const int lane = threadIdx.x & 31;
    const int wv   = threadIdx.x >> 5;
    const int row  = blockIdx.x * 8 + wv;          // 128 blocks * 8 waves = 1024

    const float4* xp = reinterpret_cast<const float4*>(x + (size_t)row * D_) + lane * 4;
    float vals[16];
    float ss = 0.f;
    #pragma unroll
    for (int j = 0; j < 4; ++j) {
        float4 q = xp[j];
        vals[4*j+0] = q.x; vals[4*j+1] = q.y; vals[4*j+2] = q.z; vals[4*j+3] = q.w;
    }
    #pragma unroll
    for (int j = 0; j < 16; ++j) ss += vals[j] * vals[j];
    ss = wave_sum32(ss);
    const float norm = sqrtf(ss);
    const float inv  = 1.0f / fmaxf(norm, 1e-12f);

    union { unsigned short h[16]; uint4 q[2]; } pk;
    #pragma unroll
    for (int j = 0; j < 16; ++j) pk.h[j] = f2bf(vals[j] * inv);
    uint4* dst = reinterpret_cast<uint4*>(xn + (size_t)row * D_ + lane * 16);
    dst[0] = pk.q[0]; dst[1] = pk.q[1];

    if (lane == 0) safe_norms[row] = fminf(fmaxf(norm, 0.001f), 100.0f);

    const int gid = blockIdx.x * blockDim.x + threadIdx.x;
    if (gid < N_) sumexp[gid] = 0.0f;               // fresh accumulator every launch
}

// ---------------- kernel 2: batch stats + per-row margins -------------------
__global__ void k_stats_margins(const float* __restrict__ safe_norms,
                                float* __restrict__ g_ang,
                                float* __restrict__ g_add) {
    __shared__ float red[256];
    __shared__ float s_mean, s_std;
    const int tid = threadIdx.x;

    float v[4], loc = 0.f;
    #pragma unroll
    for (int k = 0; k < 4; ++k) { v[k] = safe_norms[tid + k * 256]; loc += v[k]; }
    red[tid] = loc; __syncthreads();
    for (int s = 128; s > 0; s >>= 1) { if (tid < s) red[tid] += red[tid + s]; __syncthreads(); }
    if (tid == 0) s_mean = red[0] / (float)N_;
    __syncthreads();
    const float mean = s_mean;

    loc = 0.f;
    #pragma unroll
    for (int k = 0; k < 4; ++k) { float d = v[k] - mean; loc += d * d; }
    red[tid] = loc; __syncthreads();
    for (int s = 128; s > 0; s >>= 1) { if (tid < s) red[tid] += red[tid + s]; __syncthreads(); }
    if (tid == 0) s_std = sqrtf(red[0] / (float)(N_ - 1));   // unbiased (ddof=1)
    __syncthreads();
    const float stdv = s_std;

    #pragma unroll
    for (int k = 0; k < 4; ++k) {
        const int i = tid + k * 256;
        float ms = (v[k] - mean) / (stdv + EPS_) * H_SCALE;
        ms = fminf(fmaxf(ms, -1.0f), 1.0f);
        g_ang[i] = -M_MARGIN * ms;
        g_add[i] =  M_MARGIN + M_MARGIN * ms;
    }
}

// ---------------- kernel 3: bf16 WMMA GEMM + fused exp-sum ------------------
// W-stationary: each block owns CT=32 classes, stages the normalized bf16 W
// tile in LDS once (W streamed from HBM exactly once overall), then 8 waves
// (2 column strips x 4 row quarters) sweep all 1024 rows. B fragments for the
// full K=512 stay resident in VGPRs (16 x v16bf). A fragments stream from the
// WGP cache with an explicit 2-deep rotating prefetch so WMMAs only take
// partial loadcnt waits (no full drains).
__global__ void __launch_bounds__(256)
k_gemm_sumexp(const unsigned short* __restrict__ xn,
              const float* __restrict__ w,
              float* __restrict__ sumexp) {
    __shared__ unsigned short wl[CT * LSTR];   // 33,280 B
    __shared__ float lsum[N_];                 //  4,096 B

    const int tid  = threadIdx.x;
    const int lane = tid & 31;
    const int wv   = tid >> 5;
    const int c0   = blockIdx.x * CT;

    #pragma unroll
    for (int k = 0; k < 4; ++k) lsum[tid + k * 256] = 0.0f;

    // ---- stage W tile: each wave normalizes 4 class rows into LDS (bf16) ----
    for (int cc = wv * 4; cc < wv * 4 + 4; ++cc) {
        const int g = c0 + cc;
        float vals[16];
        float ss = 0.f;
        if (g < C_) {
            const float4* p = reinterpret_cast<const float4*>(w + (size_t)g * D_) + lane * 4;
            #pragma unroll
            for (int j = 0; j < 4; ++j) {
                float4 q = p[j];
                vals[4*j+0] = q.x; vals[4*j+1] = q.y; vals[4*j+2] = q.z; vals[4*j+3] = q.w;
            }
            #pragma unroll
            for (int j = 0; j < 16; ++j) ss += vals[j] * vals[j];
        } else {
            #pragma unroll
            for (int j = 0; j < 16; ++j) vals[j] = 0.f;
        }
        ss = wave_sum32(ss);
        const float inv = 1.0f / fmaxf(sqrtf(ss), 1e-12f);
        union { unsigned short h[16]; uint4 q[2]; } pk;
        #pragma unroll
        for (int j = 0; j < 16; ++j) pk.h[j] = f2bf(vals[j] * inv);
        uint4* dst = reinterpret_cast<uint4*>(&wl[cc * LSTR + lane * 16]);
        dst[0] = pk.q[0]; dst[1] = pk.q[1];
    }
    __syncthreads();

    const int strip = wv & 1;           // which 16-class strip
    const int quad  = wv >> 1;          // which 256-row quarter
    const int l16   = lane & 15;
    const int kgrp  = lane >> 4;
    const int colG  = c0 + strip * 16 + l16;
    const bool colOK = (colG < C_);

    // resident B fragments: full K = 512 for this strip
    v16bf bfr[16];
    #pragma unroll
    for (int kk = 0; kk < 16; ++kk)
        bfr[kk] = load_b_frag(&wl[(strip * 16 + l16) * LSTR + kk * 32 + kgrp * 8]);

    // A-fragment base for a row tile: row = rowBase + l16, chunk at kgrp*8
    const unsigned short* aBase = xn + (size_t)l16 * D_ + kgrp * 8;

    // 2-deep rotating prefetch, carried across row tiles
    v16bf a0 = load_a_frag(aBase + (size_t)(quad * 256) * D_);
    v16bf a1 = load_a_frag(aBase + (size_t)(quad * 256) * D_ + 32);

    for (int rt = 0; rt < 16; ++rt) {
        const int rowBase = quad * 256 + rt * 16;
        const unsigned short* ar  = aBase + (size_t)rowBase * D_;
        // next tile (dummy self-pointer on the last tile; loads stay in-bounds)
        const unsigned short* arn = aBase + (size_t)(rt < 15 ? rowBase + 16 : rowBase) * D_;

        v8f acc = {0.f, 0.f, 0.f, 0.f, 0.f, 0.f, 0.f, 0.f};
        #pragma unroll
        for (int kk = 0; kk < 16; ++kk) {
            // prefetch fragment kk+2 (rolls into the next row tile's 0/1)
            v16bf an = (kk < 14) ? load_a_frag(ar + (kk + 2) * 32)
                                 : load_a_frag(arn + (kk - 14) * 32);
            acc = __builtin_amdgcn_wmma_f32_16x16x32_bf16(
                false, a0, false, bfr[kk], (short)0, acc, false, false);
            a0 = a1; a1 = an;
        }

        // epilogue: clamp == cos(clip(acos(c))), shift-64 exp, 16-lane reduce
        // (overlaps with the prefetch of the next tile's first fragments)
        #pragma unroll
        for (int r = 0; r < 8; ++r) {
            float l = S_SCALE * fminf(fmaxf(acc[r], -COS_EPS), COS_EPS);
            float e = colOK ? __expf(l - 64.0f) : 0.0f;
            e += __shfl_xor(e, 1, 32);
            e += __shfl_xor(e, 2, 32);
            e += __shfl_xor(e, 4, 32);
            e += __shfl_xor(e, 8, 32);
            if (l16 == 0) atomicAdd(&lsum[rowBase + r + kgrp * 8], e);
        }
    }
    __syncthreads();

    #pragma unroll
    for (int k = 0; k < 4; ++k) {
        const int i = tid + k * 256;
        atomicAdd(&sumexp[i], lsum[i]);
    }
}

// ---------------- kernel 4: label-column logits (margin path) ---------------
__global__ void k_label_logits(const unsigned short* __restrict__ xn,
                               const float* __restrict__ w,
                               const int* __restrict__ labels,
                               const float* __restrict__ g_ang,
                               const float* __restrict__ g_add,
                               float* __restrict__ l_lab,
                               float* __restrict__ l_plain) {
    const int lane = threadIdx.x & 31;
    const int wv   = threadIdx.x >> 5;
    const int row  = blockIdx.x * 8 + wv;

    const int lab = labels[row];
    const float4* wp = reinterpret_cast<const float4*>(w + (size_t)lab * D_) + lane * 4;
    float vals[16];
    float ss = 0.f;
    #pragma unroll
    for (int j = 0; j < 4; ++j) {
        float4 q = wp[j];
        vals[4*j+0] = q.x; vals[4*j+1] = q.y; vals[4*j+2] = q.z; vals[4*j+3] = q.w;
    }
    #pragma unroll
    for (int j = 0; j < 16; ++j) ss += vals[j] * vals[j];
    ss = wave_sum32(ss);
    const float inv = 1.0f / fmaxf(sqrtf(ss), 1e-12f);

    const unsigned short* xr = xn + (size_t)row * D_ + lane * 16;
    float d = 0.f;
    #pragma unroll
    for (int j = 0; j < 16; ++j) {
        float wb = bf2f(f2bf(vals[j] * inv));   // round like the GEMM path
        d += bf2f(xr[j]) * wb;
    }
    d = wave_sum32(d);

    if (lane == 0) {
        float c  = fminf(fmaxf(d, -1.0f), 1.0f);
        float th = acosf(c);
        float tm = fminf(fmaxf(th + g_ang[row], EPS_), 3.14159265358979f - EPS_);
        l_lab[row]   = (cosf(tm) - g_add[row]) * S_SCALE;
        l_plain[row] = S_SCALE * fminf(fmaxf(c, -COS_EPS), COS_EPS);
    }
}

// ---------------- kernel 5: finalize loss -----------------------------------
__global__ void k_finalize(const float* __restrict__ sumexp,
                           const float* __restrict__ l_lab,
                           const float* __restrict__ l_plain,
                           float* __restrict__ out) {
    __shared__ float red[256];
    const int tid = threadIdx.x;
    float loc = 0.f;
    #pragma unroll
    for (int k = 0; k < 4; ++k) {
        const int i = tid + k * 256;
        float s = sumexp[i] + __expf(l_lab[i] - 64.0f) - __expf(l_plain[i] - 64.0f);
        s = fmaxf(s, 1e-38f);
        loc += logf(s) + 64.0f - l_lab[i];
    }
    red[tid] = loc; __syncthreads();
    for (int s = 128; s > 0; s >>= 1) { if (tid < s) red[tid] += red[tid + s]; __syncthreads(); }
    if (tid == 0) out[0] = red[0] / (float)N_;
}

// ---------------- launcher --------------------------------------------------
extern "C" void kernel_launch(void* const* d_in, const int* in_sizes, int n_in,
                              void* d_out, int out_size, void* d_ws, size_t ws_size,
                              hipStream_t stream) {
    const float* x       = (const float*)d_in[0];
    const int*   labels  = (const int*)d_in[1];
    const float* weights = (const float*)d_in[2];
    float*       out     = (float*)d_out;

    // workspace layout
    char* ws = (char*)d_ws;
    unsigned short* xn = (unsigned short*)ws;                     // 1024*512*2 = 1 MiB
    float* f          = (float*)(ws + (size_t)N_ * D_ * 2);
    float* safe_norms = f + 0 * N_;
    float* g_ang      = f + 1 * N_;
    float* g_add      = f + 2 * N_;
    float* sumexp     = f + 3 * N_;
    float* l_lab      = f + 4 * N_;
    float* l_plain    = f + 5 * N_;

    k_normalize_x  <<<N_ / 8, 256, 0, stream>>>(x, xn, safe_norms, sumexp);
    k_stats_margins<<<1,      256, 0, stream>>>(safe_norms, g_ang, g_add);
    k_gemm_sumexp  <<<GEMM_BLOCKS, 256, 0, stream>>>(xn, weights, sumexp);
    k_label_logits <<<N_ / 8, 256, 0, stream>>>(xn, weights, labels, g_ang, g_add, l_lab, l_plain);
    k_finalize     <<<1,      256, 0, stream>>>(sumexp, l_lab, l_plain, out);
}
